// AxialAttention_25984552141241
// MI455X (gfx1250) — matrix operand order repaired and checked by reading, not compiled, not run on hardware
//
#include <hip/hip_runtime.h>

typedef __bf16 bf16;
typedef __attribute__((ext_vector_type(16))) __bf16 v16bf;
typedef __attribute__((ext_vector_type(8)))  __bf16 v8bf;
typedef __attribute__((ext_vector_type(8)))  float  v8f;

constexpr int Bn = 8, Hn = 128, Wn = 128, Cn = 256, Dn = 256;

union V16U { v16bf v; v8bf h[2]; };
union BF4  { bf16 b[4]; uint2 u2; };

// native f32 -> bf16 (RNE); clang lowers to v_cvt_pk_bf16_f32 / bf16 cvt on gfx1250
__device__ __forceinline__ bf16 f2bf(float f) { return (bf16)f; }

// A-fragment (16x32 bf16): lane m holds K = k0 + hf*8 + {0..7} and k0+16+hf*8+{0..7}
__device__ __forceinline__ v16bf load_a(const bf16* base, int ld, int row0, int k0, int lane) {
  const int m = lane & 15, hf = lane >> 4;
  const bf16* p = base + (row0 + m) * ld + k0 + hf * 8;
  V16U u;
  u.h[0] = *(const v8bf*)(p);
  u.h[1] = *(const v8bf*)(p + 16);
  return u.v;
}

// B-fragment (32x16 bf16) from a matrix stored with rows = N, cols = K (i.e. B^T row-major)
__device__ __forceinline__ v16bf load_b(const bf16* bt, int ld, int n0, int k0, int lane) {
  const int n = lane & 15, hf = lane >> 4;
  const bf16* p = bt + (n0 + n) * ld + k0 + hf * 8;
  V16U u;
  u.h[0] = *(const v8bf*)(p);
  u.h[1] = *(const v8bf*)(p + 16);
  return u.v;
}

// C/D layout: VGPR r, lane -> (M = r + 8*(lane/16), N = lane%16)
__device__ __forceinline__ void store_d(bf16* dst, int ld, int row0, int n0, v8f acc, int lane) {
  const int n = lane & 15, hf = lane >> 4;
  bf16* p = dst + (row0 + 8 * hf) * ld + n0 + n;
#pragma unroll
  for (int r = 0; r < 8; ++r) p[r * ld] = f2bf(acc[r]);
}

// store transposed: dst[(n0+n)*ld + col0 + r + 8*hf]  (one contiguous 16B store per lane)
__device__ __forceinline__ void store_d_T(bf16* dst, int ld, int col0, int n0, v8f acc, int lane) {
  const int n = lane & 15, hf = lane >> 4;
  bf16* p = dst + (n0 + n) * ld + col0 + 8 * hf;
  v8bf t;
#pragma unroll
  for (int r = 0; r < 8; ++r) t[r] = f2bf(acc[r]);
  *(v8bf*)p = t;
}

#define WMMA_BF16(a, b, c) \
  __builtin_amdgcn_wmma_f32_16x16x32_bf16(false, (a), false, (b), (short)0, (c), false, false)

__global__ void prep_weights(const float* __restrict__ wq, const float* __restrict__ wk,
                             const float* __restrict__ wv, bf16* __restrict__ ws) {
  int i = blockIdx.x * blockDim.x + threadIdx.x;
  if (i >= Cn * Dn) return;
  int cc = i / Dn, dd = i % Dn;
  bf16* wqT = ws;
  bf16* wkT = ws + Cn * Dn;
  bf16* wvT = ws + 2 * Cn * Dn;
  bf16* wvb = ws + 3 * Cn * Dn;
  wqT[dd * Cn + cc] = f2bf(wq[i]);
  wkT[dd * Cn + cc] = f2bf(wk[i]);
  wvT[dd * Cn + cc] = f2bf(wv[i]);
  wvb[i]            = f2bf(wv[i]);
}

__global__ __launch_bounds__(256) void axial_attn(const float* __restrict__ x,
                                                  const bf16* __restrict__ wqT,
                                                  const bf16* __restrict__ wkT,
                                                  const bf16* __restrict__ wvT,
                                                  const bf16* __restrict__ wvb,
                                                  float* __restrict__ out) {
  __shared__ __align__(16) bf16 lds_xq[Wn * Cn];      // X -> Q -> attended (per-wave strips)
  __shared__ __align__(16) bf16 lds_k [Wn * Dn];      // K row-major [w][d]
  __shared__ __align__(16) bf16 lds_vt[Dn * Wn];      // V^T [d][w]
  __shared__ __align__(16) bf16 lds_p [8 * 16 * Wn];  // per-wave probs [16][W]

  const int bh    = blockIdx.x;
  const int tid   = threadIdx.x;
  const int lane  = tid & 31;
  const int wave  = tid >> 5;
  const int mBase = wave * 16;

  // ---- Phase 0: load X tile (fp32, coalesced) -> bf16 LDS
  const float* xt = x + (size_t)bh * (Wn * Cn);
  for (int i = tid; i < (Wn * Cn) / 4; i += 256) {
    float4 f = ((const float4*)xt)[i];
    BF4 t;
    t.b[0] = f2bf(f.x); t.b[1] = f2bf(f.y); t.b[2] = f2bf(f.z); t.b[3] = f2bf(f.w);
    *(uint2*)&lds_xq[4 * i] = t.u2;
  }
  __syncthreads();

  // ---- Phase 1: projections; each wave owns rows [mBase, mBase+16)
  v16bf ax[8];
#pragma unroll
  for (int kb = 0; kb < 8; ++kb) ax[kb] = load_a(lds_xq, Cn, mBase, kb * 32, lane);

#pragma unroll 1
  for (int nt = 0; nt < 16; ++nt) {                    // K
    v8f acc = {};
#pragma unroll
    for (int kb = 0; kb < 8; ++kb)
      acc = WMMA_BF16(ax[kb], load_b(wkT, Cn, nt * 16, kb * 32, lane), acc);
    store_d(lds_k, Dn, mBase, nt * 16, acc, lane);
  }
#pragma unroll 1
  for (int nt = 0; nt < 16; ++nt) {                    // V -> transposed
    v8f acc = {};
#pragma unroll
    for (int kb = 0; kb < 8; ++kb)
      acc = WMMA_BF16(ax[kb], load_b(wvT, Cn, nt * 16, kb * 32, lane), acc);
    store_d_T(lds_vt, Wn, mBase, nt * 16, acc, lane);
  }
  v8f qacc[16];                                        // Q held in registers
#pragma unroll
  for (int nt = 0; nt < 16; ++nt) {
    qacc[nt] = {};
#pragma unroll
    for (int kb = 0; kb < 8; ++kb)
      qacc[nt] = WMMA_BF16(ax[kb], load_b(wqT, Cn, nt * 16, kb * 32, lane), qacc[nt]);
  }
#pragma unroll
  for (int nt = 0; nt < 16; ++nt)                      // overwrite own X strip with Q
    store_d(lds_xq, Cn, mBase, nt * 16, qacc[nt], lane);
  __syncthreads();

  // ---- Phase 2: S = Q K^T, softmax over 128 keys
  v16bf aq[8];
#pragma unroll
  for (int kb = 0; kb < 8; ++kb) aq[kb] = load_a(lds_xq, Cn, mBase, kb * 32, lane);
  v8f sacc[8];
#pragma unroll
  for (int nt = 0; nt < 8; ++nt) {
    sacc[nt] = {};
#pragma unroll
    for (int kb = 0; kb < 8; ++kb)
      sacc[nt] = WMMA_BF16(aq[kb], load_b(lds_k, Dn, nt * 16, kb * 32, lane), sacc[nt]);
  }
  // row softmax: each row spans the 8 register slots x 16 lanes of one half-wave
#pragma unroll
  for (int r = 0; r < 8; ++r) {
    float mx = sacc[0][r];
#pragma unroll
    for (int nt = 1; nt < 8; ++nt) mx = fmaxf(mx, sacc[nt][r]);
#pragma unroll
    for (int off = 8; off > 0; off >>= 1) mx = fmaxf(mx, __shfl_xor(mx, off, 32));
    float sum = 0.f;
#pragma unroll
    for (int nt = 0; nt < 8; ++nt) { float e = __expf(sacc[nt][r] - mx); sacc[nt][r] = e; sum += e; }
#pragma unroll
    for (int off = 8; off > 0; off >>= 1) sum += __shfl_xor(sum, off, 32);
    float inv = 1.0f / sum;
#pragma unroll
    for (int nt = 0; nt < 8; ++nt) sacc[nt][r] *= inv;
  }
  bf16* myP = lds_p + wave * (16 * Wn);
#pragma unroll
  for (int nt = 0; nt < 8; ++nt) store_d(myP, Wn, 0, nt * 16, sacc[nt], lane);

  // ---- Phase 2.5: attended = P V (overwrites own Q strip)
  v16bf ap[4];
#pragma unroll
  for (int kb = 0; kb < 4; ++kb) ap[kb] = load_a(myP, Wn, 0, kb * 32, lane);
#pragma unroll 1
  for (int dt = 0; dt < 16; ++dt) {
    v8f acc = {};
#pragma unroll
    for (int kb = 0; kb < 4; ++kb)
      acc = WMMA_BF16(ap[kb], load_b(lds_vt, Wn, dt * 16, kb * 32, lane), acc);
    store_d(lds_xq, Cn, mBase, dt * 16, acc, lane);
  }

  // ---- Phase 3: out = attended @ Wv^T -> global fp32
  v16bf aa[8];
#pragma unroll
  for (int kb = 0; kb < 8; ++kb) aa[kb] = load_a(lds_xq, Cn, mBase, kb * 32, lane);
  float* outt = out + (size_t)bh * (Wn * Cn);
  const int n = lane & 15, hf = lane >> 4;
#pragma unroll 1
  for (int nt = 0; nt < 16; ++nt) {
    v8f acc = {};
#pragma unroll
    for (int kb = 0; kb < 8; ++kb)
      acc = WMMA_BF16(aa[kb], load_b(wvb, Dn, nt * 16, kb * 32, lane), acc);
    float* p = outt + (size_t)(mBase + 8 * hf) * Cn + nt * 16 + n;
#pragma unroll
    for (int r = 0; r < 8; ++r) p[r * Cn] = acc[r];
  }
}

extern "C" void kernel_launch(void* const* d_in, const int* in_sizes, int n_in,
                              void* d_out, int out_size, void* d_ws, size_t ws_size,
                              hipStream_t stream) {
  (void)in_sizes; (void)n_in; (void)out_size; (void)ws_size;
  const float* x  = (const float*)d_in[0];
  const float* wq = (const float*)d_in[1];
  const float* wk = (const float*)d_in[2];
  const float* wv = (const float*)d_in[3];
  bf16* ws = (bf16*)d_ws;   // needs 4 * 256*256 * 2B = 512 KB
  prep_weights<<<(Cn * Dn + 255) / 256, 256, 0, stream>>>(wq, wk, wv, ws);
  axial_attn<<<Bn * Hn, 256, 0, stream>>>(x, ws, ws + Cn * Dn, ws + 2 * Cn * Dn,
                                          ws + 3 * Cn * Dn, (float*)d_out);
}
